// PointTransformer_FPMOD_75642964017685
// MI455X (gfx1250) — matrix operand order repaired and checked by reading, not compiled, not run on hardware
//
#include <hip/hip_runtime.h>

// ---------------------------------------------------------------------------
// Types for CDNA5 WMMA
// ---------------------------------------------------------------------------
typedef __attribute__((ext_vector_type(16))) _Float16 v16h;
typedef __attribute__((ext_vector_type(8)))  _Float16 v8h;
typedef __attribute__((ext_vector_type(4)))  _Float16 v4h;
typedef __attribute__((ext_vector_type(8)))  float    v8f;

#define EPS_BN 1e-5f

// ---------------------------------------------------------------------------
// Fragment loaders (contiguous k = 16*(lane/16) + s order, same in A and B), built from ds_load_b128 pairs.
//  A tile  : row-major [m][k], row stride 40 halves (80B -> conflict-free b128)
//  B tile  : col-major [n][k], row stride btStride halves (multiple of 8)
// ---------------------------------------------------------------------------
__device__ __forceinline__ v16h load_afrag(const _Float16* As, int wm, int lane) {
    const int arow = wm * 16 + (lane & 15);
    const int kA   = (lane < 16) ? 0 : 16;  // halves h -> K kA+h
    const _Float16* p = As + arow * 40 + kA;
    v8h lo = *(const v8h*)p;
    v8h hi = *(const v8h*)(p + 8);
    return __builtin_shufflevector(lo, hi, 0, 1, 2, 3, 4, 5, 6, 7, 8, 9, 10, 11, 12, 13, 14, 15);
}

__device__ __forceinline__ v16h load_bfrag(const _Float16* Bt, int btStride, int ncol, int lane) {
    const int bcol = ncol + (lane & 15);
    const int kB   = (lane < 16) ? 0 : 16;  // halves h -> K kB+h
    const _Float16* p = Bt + (long)bcol * btStride + kB;
    v8h lo = *(const v8h*)p;
    v8h hi = *(const v8h*)(p + 8);
    return __builtin_shufflevector(lo, hi, 0, 1, 2, 3, 4, 5, 6, 7, 8, 9, 10, 11, 12, 13, 14, 15);
}

// ---------------------------------------------------------------------------
// Generic batched GEMM:  Y[b,m,n] = sum_k A[(b),(m,k)|(k,m)] * X[b,k,n] (+bias[m])
//   PBA: A is per-batch (stride strideA), else shared weight
//   TA : A indexed transposed A[k*lda+m], else A[m*lda+k]
// block = 256 threads (8 waves arranged 4x2), tile 64(M) x 64(N), K step 32.
// Each wave: one A fragment reused across two B fragments -> 2 WMMA / K-step.
// ---------------------------------------------------------------------------
template <bool PBA, bool TA>
__global__ __launch_bounds__(256) void gemm_wmma(const float* __restrict__ A,
                                                 const float* __restrict__ X,
                                                 const float* __restrict__ bias,
                                                 float* __restrict__ Y,
                                                 int M, int N, int K, int lda, long strideA) {
    __shared__ _Float16 As[64 * 40] __attribute__((aligned(16)));
    __shared__ _Float16 Bt[64 * 40] __attribute__((aligned(16)));
    const int bz = blockIdx.z;
    const float* Ab = A + (PBA ? (long)bz * strideA : 0);
    const float* Xb = X + (long)bz * K * N;
    float* Yb = Y + (long)bz * M * N;
    const int m0 = blockIdx.y * 64, n0 = blockIdx.x * 64;
    const int tid = threadIdx.x, lane = tid & 31, wv = tid >> 5, wm = wv & 3, wn = wv >> 2;
    v8f acc0 = {}, acc1 = {};
    for (int kt = 0; kt < K; kt += 32) {
        // ---- stage A tile (64 x 32) ----
        if (!TA && (m0 + 64 <= M) && (kt + 32 <= K)) {
            for (int t = tid; t < 512; t += 256) {     // 64*32/4 float4 chunks
                int r = t >> 3, q4 = (t & 7) * 4;
                const float4 f = *(const float4*)(Ab + (long)(m0 + r) * lda + kt + q4);
                v4h h = { (_Float16)f.x, (_Float16)f.y, (_Float16)f.z, (_Float16)f.w };
                *(v4h*)(As + r * 40 + q4) = h;
            }
        } else {
            for (int t = tid; t < 64 * 32; t += 256) {
                int r = t >> 5, q = t & 31;
                int m = m0 + r, k = kt + q;
                float v = 0.f;
                if (m < M && k < K) v = TA ? Ab[(long)k * lda + m] : Ab[(long)m * lda + k];
                As[r * 40 + q] = (_Float16)v;
            }
        }
        // ---- stage X tile (32 x 64), stored transposed [n][k] ----
        if (kt + 32 <= K) {
            for (int t = tid; t < 512; t += 256) {     // 32*64/4 float4 chunks along n
                int r = t >> 4, q4 = (t & 15) * 4;     // r = k-local, q4 = n-local
                const float4 f = *(const float4*)(Xb + (long)(kt + r) * N + n0 + q4);
                Bt[(q4 + 0) * 40 + r] = (_Float16)f.x;
                Bt[(q4 + 1) * 40 + r] = (_Float16)f.y;
                Bt[(q4 + 2) * 40 + r] = (_Float16)f.z;
                Bt[(q4 + 3) * 40 + r] = (_Float16)f.w;
            }
        } else {
            for (int t = tid; t < 32 * 64; t += 256) {
                int r = t >> 6, q = t & 63;
                int k = kt + r, n = n0 + q;
                float v = 0.f;
                if (k < K && n < N) v = Xb[(long)k * N + n];
                Bt[q * 40 + r] = (_Float16)v;
            }
        }
        // prefetch next K tile into L2 (global_prefetch_b8)
        if (kt + 32 < K) {
            if (tid < 64) {
                int m = m0 + tid;
                if (!TA && m < M) __builtin_prefetch(Ab + (long)m * lda + kt + 32, 0, 1);
            } else if (tid < 96) {
                __builtin_prefetch(Xb + (long)(kt + 32 + (tid - 64)) * N + n0, 0, 1);
            }
        }
        __syncthreads();
        v16h af = load_afrag(As, wm, lane);
        v16h b0 = load_bfrag(Bt, 40, wn * 32 + 0, lane);
        v16h b1 = load_bfrag(Bt, 40, wn * 32 + 16, lane);
        acc0 = __builtin_amdgcn_wmma_f32_16x16x32_f16(false, af, false, b0, (short)0, acc0, false, false);
        acc1 = __builtin_amdgcn_wmma_f32_16x16x32_f16(false, af, false, b1, (short)0, acc1, false, false);
        __syncthreads();
    }
    const int colbase = n0 + wn * 32 + (lane & 15);
#pragma unroll
    for (int v = 0; v < 8; ++v) {
        int row = m0 + wm * 16 + v + ((lane < 16) ? 0 : 8);
        if (row < M) {
            float bia = bias ? bias[row] : 0.f;
            if (colbase < N)      Yb[(long)row * N + colbase]      = acc0[v] + bia;
            if (colbase + 16 < N) Yb[(long)row * N + colbase + 16] = acc1[v] + bia;
        }
    }
}

// ---------------------------------------------------------------------------
// BN helpers (batch statistics from raw sum / sumsq)
// ---------------------------------------------------------------------------
__device__ __forceinline__ float bn_relu_val(float v, const float* st, const float* g,
                                             const float* bt, int ch, float invc) {
    float m  = st[2 * ch] * invc;
    float vr = st[2 * ch + 1] * invc - m * m;
    return fmaxf(0.f, g[ch] * (v - m) * rsqrtf(vr + EPS_BN) + bt[ch]);
}

__global__ __launch_bounds__(256) void bn_stats(const float* __restrict__ x, float* __restrict__ stats,
                                                int C, int Bn, int Nn) {
    int c = blockIdx.x, tid = threadIdx.x;
    __shared__ float rs[256], rq[256];
    float s = 0.f, q = 0.f;
    int tot = Bn * Nn;
    for (int t = tid; t < tot; t += 256) {
        int b = t / Nn, i = t - b * Nn;
        float v = x[((long)b * C + c) * Nn + i];
        s += v; q += v * v;
    }
    rs[tid] = s; rq[tid] = q; __syncthreads();
    for (int k = 128; k > 0; k >>= 1) {
        if (tid < k) { rs[tid] += rs[tid + k]; rq[tid] += rq[tid + k]; }
        __syncthreads();
    }
    if (tid == 0) { stats[2 * c] = rs[0]; stats[2 * c + 1] = rq[0]; }
}

// act: 0=none 1=relu 2=leaky(0.2); optional residual add after activation
__global__ __launch_bounds__(256) void bn_apply(const float* __restrict__ x, float* __restrict__ y,
                                                const float* __restrict__ stats,
                                                const float* __restrict__ g, const float* __restrict__ bt,
                                                const float* __restrict__ res,
                                                int C, int Nn, long total, int act, float invc) {
    long t = (long)blockIdx.x * blockDim.x + threadIdx.x;
    if (t >= total) return;
    int c = (int)((t / Nn) % C);
    float m  = stats[2 * c] * invc;
    float vr = stats[2 * c + 1] * invc - m * m;
    float v  = g[c] * (x[t] - m) * rsqrtf(vr + EPS_BN) + bt[c];
    if (act == 1) v = fmaxf(v, 0.f);
    else if (act == 2) v = (v < 0.f) ? 0.2f * v : v;
    if (res) v += res[t];
    y[t] = v;
}

// ---------------------------------------------------------------------------
// Farthest point sampling: one block per batch, serial over npoint iterations.
// ---------------------------------------------------------------------------
__global__ __launch_bounds__(256) void fps_kernel(const float* __restrict__ xyz, int* __restrict__ idx,
                                                  int N, int np) {
    int b = blockIdx.x, tid = threadIdx.x;
    __shared__ float dist[2048];
    __shared__ float rv[256];
    __shared__ int   ri[256];
    __shared__ int   sFar;
    for (int i = tid; i < N; i += 256) dist[i] = 1e10f;
    if (tid == 0) sFar = 0;
    __syncthreads();
    for (int it = 0; it < np; ++it) {
        int far = sFar;
        if (tid == 0) idx[b * np + it] = far;
        float cx = xyz[((long)b * N + far) * 3 + 0];
        float cy = xyz[((long)b * N + far) * 3 + 1];
        float cz = xyz[((long)b * N + far) * 3 + 2];
        float bv = -1.f; int bi = 0;
        for (int i = tid; i < N; i += 256) {
            float dx = xyz[((long)b * N + i) * 3 + 0] - cx;
            float dy = xyz[((long)b * N + i) * 3 + 1] - cy;
            float dz = xyz[((long)b * N + i) * 3 + 2] - cz;
            float d = dx * dx + dy * dy + dz * dz;
            float nd = fminf(dist[i], d);
            dist[i] = nd;
            if (nd > bv) { bv = nd; bi = i; }
        }
        rv[tid] = bv; ri[tid] = bi; __syncthreads();
        for (int s = 128; s > 0; s >>= 1) {
            if (tid < s && rv[tid + s] > rv[tid]) { rv[tid] = rv[tid + s]; ri[tid] = ri[tid + s]; }
            __syncthreads();
        }
        if (tid == 0) sFar = ri[0];
        __syncthreads();
    }
}

__global__ __launch_bounds__(256) void gather_xyz(const float* __restrict__ xyz, const int* __restrict__ idx,
                                                  float* __restrict__ o, int nsrc, int np, int total) {
    int t = blockIdx.x * blockDim.x + threadIdx.x;
    if (t >= total) return;
    int b = t / np;
    int j = idx[t];
#pragma unroll
    for (int k = 0; k < 3; ++k) o[(long)t * 3 + k] = xyz[((long)b * nsrc + j) * 3 + k];
}

// KNN top-32 (smallest distances): one block per query, 32 masked argmins.
__global__ __launch_bounds__(256) void knn32_kernel(const float* __restrict__ q, const float* __restrict__ src,
                                                    int* __restrict__ out, int ns) {
    int b = blockIdx.y, qi = blockIdx.x, nq = gridDim.x, tid = threadIdx.x;
    __shared__ float d[2048];
    __shared__ float rv[256];
    __shared__ int   ri[256];
    float qx = q[((long)b * nq + qi) * 3 + 0];
    float qy = q[((long)b * nq + qi) * 3 + 1];
    float qz = q[((long)b * nq + qi) * 3 + 2];
    for (int i = tid; i < ns; i += 256) {
        float dx = src[((long)b * ns + i) * 3 + 0] - qx;
        float dy = src[((long)b * ns + i) * 3 + 1] - qy;
        float dz = src[((long)b * ns + i) * 3 + 2] - qz;
        d[i] = dx * dx + dy * dy + dz * dz;
    }
    __syncthreads();
    for (int k = 0; k < 32; ++k) {
        float bv = 3e38f; int bi = 0;
        for (int i = tid; i < ns; i += 256)
            if (d[i] < bv) { bv = d[i]; bi = i; }
        rv[tid] = bv; ri[tid] = bi; __syncthreads();
        for (int s = 128; s > 0; s >>= 1) {
            if (tid < s && rv[tid + s] < rv[tid]) { rv[tid] = rv[tid + s]; ri[tid] = ri[tid + s]; }
            __syncthreads();
        }
        if (tid == 0) { out[((long)(b * nq + qi)) * 32 + k] = ri[0]; d[ri[0]] = 3e38f; }
        __syncthreads();
    }
}

// ---------------------------------------------------------------------------
// Fused local_op (sample_and_group feature build + 2x (conv+BN+relu) + maxpool)
// One block per group. MODE 0: stage-1 stats. MODE 1: stage-2 stats.
// MODE 2: full pipeline + max over the 32 samples.
// Group matrix (2d x 32) is gathered into LDS transposed [s][k] (stride c+8)
// so WMMA B fragments are contiguous b128 loads; BN stats via f32 atomics.
// ---------------------------------------------------------------------------
template <int MODE>
__global__ __launch_bounds__(256) void local_pass(const float* __restrict__ feat,
                                                  const int* __restrict__ fpsIdx,
                                                  const int* __restrict__ knn,
                                                  const float* __restrict__ W1, const float* __restrict__ W2,
                                                  float* __restrict__ stats1, float* __restrict__ stats2,
                                                  const float* __restrict__ g1, const float* __restrict__ b1,
                                                  const float* __restrict__ g2, const float* __restrict__ b2,
                                                  float* __restrict__ out,
                                                  int d, int c, int np, int nsrc, float invc) {
    __shared__ _Float16 sX[32 * 520] __attribute__((aligned(16)));   // [s][k], stride c+8
    __shared__ _Float16 sY[32 * 520] __attribute__((aligned(16)));
    __shared__ _Float16 sW[64 * 40]  __attribute__((aligned(16)));
    __shared__ float    sMax[64 * 32];
    __shared__ int      sIdx[33];
    const int b = blockIdx.y, g = blockIdx.x, tid = threadIdx.x;
    const int cs = c + 8;
    if (tid < 32)       sIdx[tid] = knn[((long)(b * np + g)) * 32 + tid];
    else if (tid == 32) sIdx[32]  = fpsIdx[b * np + g];
    __syncthreads();
    const int ctr = sIdx[32];
    for (int t = tid; t < c * 32; t += 256) {
        int r = t >> 5, s = t & 31;
        float v;
        if (r < d) v = feat[((long)b * d + r) * nsrc + sIdx[s]] - feat[((long)b * d + r) * nsrc + ctr];
        else       v = feat[((long)b * d + (r - d)) * nsrc + ctr];
        sX[s * cs + r] = (_Float16)v;
    }
    __syncthreads();

    const int lane = tid & 31, wv = tid >> 5, wm = wv & 3, wn = wv >> 2;
    const int nMB = c >> 6;

    // ---- stage 1: y1 = W1 * xloc ----
    for (int mb = 0; mb < nMB; ++mb) {
        v8f acc = {};
        for (int kt = 0; kt < c; kt += 32) {
            for (int t = tid; t < 512; t += 256) {        // 64x32 weights, float4 chunks
                int r = t >> 3, q4 = (t & 7) * 4;
                const float4 f = *(const float4*)(W1 + (long)(mb * 64 + r) * c + kt + q4);
                v4h h = { (_Float16)f.x, (_Float16)f.y, (_Float16)f.z, (_Float16)f.w };
                *(v4h*)(sW + r * 40 + q4) = h;
            }
            if (kt + 32 < c && tid < 64)
                __builtin_prefetch(W1 + (long)(mb * 64 + tid) * c + kt + 32, 0, 1);
            __syncthreads();
            v16h af = load_afrag(sW, wm, lane);
            v16h bf = load_bfrag(sX + kt, cs, wn * 16, lane);
            acc = __builtin_amdgcn_wmma_f32_16x16x32_f16(false, af, false, bf, (short)0, acc, false, false);
            __syncthreads();
        }
        const int col = wn * 16 + (lane & 15);
#pragma unroll
        for (int v = 0; v < 8; ++v) {
            int row = mb * 64 + wm * 16 + v + ((lane < 16) ? 0 : 8);
            float val = acc[v];
            if constexpr (MODE == 0) {
                atomicAdd(&stats1[2 * row], val);
                atomicAdd(&stats1[2 * row + 1], val * val);
            } else {
                sY[col * cs + row] = (_Float16)bn_relu_val(val, stats1, g1, b1, row, invc);
            }
        }
    }
    if constexpr (MODE == 0) return;
    __syncthreads();

    // ---- stage 2: y2 = W2 * relu(bn1(y1)) ----
    for (int mb = 0; mb < nMB; ++mb) {
        v8f acc = {};
        for (int kt = 0; kt < c; kt += 32) {
            for (int t = tid; t < 512; t += 256) {
                int r = t >> 3, q4 = (t & 7) * 4;
                const float4 f = *(const float4*)(W2 + (long)(mb * 64 + r) * c + kt + q4);
                v4h h = { (_Float16)f.x, (_Float16)f.y, (_Float16)f.z, (_Float16)f.w };
                *(v4h*)(sW + r * 40 + q4) = h;
            }
            if (kt + 32 < c && tid < 64)
                __builtin_prefetch(W2 + (long)(mb * 64 + tid) * c + kt + 32, 0, 1);
            __syncthreads();
            v16h af = load_afrag(sW, wm, lane);
            v16h bf = load_bfrag(sY + kt, cs, wn * 16, lane);
            acc = __builtin_amdgcn_wmma_f32_16x16x32_f16(false, af, false, bf, (short)0, acc, false, false);
            __syncthreads();
        }
        const int col = wn * 16 + (lane & 15);
        if constexpr (MODE == 1) {
#pragma unroll
            for (int v = 0; v < 8; ++v) {
                int row = mb * 64 + wm * 16 + v + ((lane < 16) ? 0 : 8);
                float val = acc[v];
                atomicAdd(&stats2[2 * row], val);
                atomicAdd(&stats2[2 * row + 1], val * val);
            }
        } else {
#pragma unroll
            for (int v = 0; v < 8; ++v) {
                int rloc = wm * 16 + v + ((lane < 16) ? 0 : 8);
                int row  = mb * 64 + rloc;
                sMax[rloc * 32 + col] = bn_relu_val(acc[v], stats2, g2, b2, row, invc);
            }
            __syncthreads();
            if (tid < 64) {
                float mx = -3e38f;
                for (int s = 0; s < 32; ++s) mx = fmaxf(mx, sMax[tid * 32 + s]);
                out[((long)b * c + mb * 64 + tid) * np + g] = mx;
            }
            __syncthreads();
        }
    }
}

// ---------------------------------------------------------------------------
// Attention softmax (rows) and column renormalization
// ---------------------------------------------------------------------------
__global__ __launch_bounds__(256) void softmax_rows(float* __restrict__ e, int n) {
    long row = blockIdx.x;
    float* p = e + row * (long)n;
    int tid = threadIdx.x;
    __shared__ float red[256];
    float v = (tid < n) ? p[tid] : -3e38f;
    red[tid] = v; __syncthreads();
    for (int s = 128; s > 0; s >>= 1) { if (tid < s) red[tid] = fmaxf(red[tid], red[tid + s]); __syncthreads(); }
    float mx = red[0]; __syncthreads();
    float ev = (tid < n) ? __expf(v - mx) : 0.f;
    red[tid] = ev; __syncthreads();
    for (int s = 128; s > 0; s >>= 1) { if (tid < s) red[tid] += red[tid + s]; __syncthreads(); }
    if (tid < n) p[tid] = ev / red[0];
}

__global__ __launch_bounds__(256) void colnorm(float* __restrict__ e, int n) {
    int b = blockIdx.x / n, m = blockIdx.x % n, tid = threadIdx.x;
    float* base = e + (long)b * n * n;
    __shared__ float red[256];
    float s = 0.f;
    for (int r = tid; r < n; r += 256) s += base[(long)r * n + m];
    red[tid] = s; __syncthreads();
    for (int k = 128; k > 0; k >>= 1) { if (tid < k) red[tid] += red[tid + k]; __syncthreads(); }
    float inv = 1.0f / (1e-9f + red[0]);
    for (int r = tid; r < n; r += 256) base[(long)r * n + m] *= inv;
}

// ---------------------------------------------------------------------------
// Misc elementwise / gather kernels
// ---------------------------------------------------------------------------
__global__ __launch_bounds__(256) void vsub(const float* a, const float* b, float* o, long n) {
    long t = (long)blockIdx.x * blockDim.x + threadIdx.x;
    if (t < n) o[t] = a[t] - b[t];
}

__global__ __launch_bounds__(256) void copy_rows(const float* __restrict__ src, float* __restrict__ dst,
                                                 int ro, int C, int Ctot, int Nn, long total) {
    long t = (long)blockIdx.x * blockDim.x + threadIdx.x;
    if (t >= total) return;
    int n = (int)(t % Nn);
    long bc = t / Nn;
    int c = (int)(bc % C), b = (int)(bc / C);
    dst[((long)b * Ctot + ro + c) * Nn + n] = src[t];
}

__global__ __launch_bounds__(256) void maxn_kernel(const float* __restrict__ x, float* __restrict__ mx,
                                                   int Nn, int total) {
    int t = blockIdx.x * blockDim.x + threadIdx.x;
    if (t >= total) return;
    const float* p = x + (long)t * Nn;
    float m = -3e38f;
    for (int i = 0; i < Nn; ++i) m = fmaxf(m, p[i]);
    mx[t] = m;
}

__global__ __launch_bounds__(256) void buildfuse(const float* __restrict__ outs, const float* __restrict__ hmax,
                                                 float* __restrict__ y, int Nn, long total) {
    long t = (long)blockIdx.x * blockDim.x + threadIdx.x;
    if (t >= total) return;
    int n = (int)(t % Nn);
    long bc = t / Nn;
    int c = (int)(bc % 4096), b = (int)(bc / 4096);
    y[t] = (c < 2048) ? outs[((long)b * 2048 + c) * Nn + n] : hmax[b * 2048 + (c - 2048)];
}

// 3-NN + inverse-distance weights for feature propagation (one thread/query)
__global__ __launch_bounds__(256) void knn3w(const float* __restrict__ q, const float* __restrict__ s,
                                             int* __restrict__ idx3, float* __restrict__ w3,
                                             int total, int nq, int ns) {
    int t = blockIdx.x * blockDim.x + threadIdx.x;
    if (t >= total) return;
    int b = t / nq, i = t % nq;
    float qx = q[((long)b * nq + i) * 3 + 0];
    float qy = q[((long)b * nq + i) * 3 + 1];
    float qz = q[((long)b * nq + i) * 3 + 2];
    float d0 = 3e38f, d1 = 3e38f, d2 = 3e38f;
    int i0 = 0, i1 = 0, i2 = 0;
    for (int j = 0; j < ns; ++j) {
        float dx = s[((long)b * ns + j) * 3 + 0] - qx;
        float dy = s[((long)b * ns + j) * 3 + 1] - qy;
        float dz = s[((long)b * ns + j) * 3 + 2] - qz;
        float d = dx * dx + dy * dy + dz * dz;
        if (d < d0)      { d2 = d1; i2 = i1; d1 = d0; i1 = i0; d0 = d; i0 = j; }
        else if (d < d1) { d2 = d1; i2 = i1; d1 = d;  i1 = j; }
        else if (d < d2) { d2 = d;  i2 = j; }
    }
    float r0 = 1.f / (d0 + 1e-8f), r1 = 1.f / (d1 + 1e-8f), r2 = 1.f / (d2 + 1e-8f);
    float rs = r0 + r1 + r2;
    idx3[(long)t * 3 + 0] = i0; idx3[(long)t * 3 + 1] = i1; idx3[(long)t * 3 + 2] = i2;
    w3[(long)t * 3 + 0] = r0 / rs; w3[(long)t * 3 + 1] = r1 / rs; w3[(long)t * 3 + 2] = r2 / rs;
}

// Build [points1 ; interp(points2)] concat buffer for feat_prop
__global__ __launch_bounds__(256) void fp_concat(const float* __restrict__ p1, const float* __restrict__ p2,
                                                 const int* __restrict__ idx3, const float* __restrict__ w3,
                                                 float* __restrict__ out, int C1, int C2, int nd, int ns,
                                                 long total) {
    long t = (long)blockIdx.x * blockDim.x + threadIdx.x;
    if (t >= total) return;
    int Ct = C1 + C2;
    int n = (int)(t % nd);
    long bc = t / nd;
    int cch = (int)(bc % Ct), b = (int)(bc / Ct);
    float v;
    if (cch < C1) {
        v = p1[((long)b * C1 + cch) * nd + n];
    } else {
        int c2 = cch - C1;
        long base = ((long)b * nd + n) * 3;
        v = 0.f;
#pragma unroll
        for (int k = 0; k < 3; ++k)
            v += w3[base + k] * p2[((long)b * C2 + c2) * ns + idx3[base + k]];
    }
    out[t] = v;
}

__global__ __launch_bounds__(256) void transpose_in(const float* __restrict__ x, float* __restrict__ y,
                                                    int Nn, long total) {
    long t = (long)blockIdx.x * blockDim.x + threadIdx.x;
    if (t >= total) return;
    int n = (int)(t % Nn);
    long bk = t / Nn;
    int k = (int)(bk % 3), b = (int)(bk / 3);
    y[t] = x[((long)b * Nn + n) * 3 + k];
}

__global__ __launch_bounds__(256) void transpose_out(const float* __restrict__ lg, float* __restrict__ o,
                                                     int Nn, long total) {
    long t = (long)blockIdx.x * blockDim.x + threadIdx.x;
    if (t >= total) return;
    int c = (int)(t % 8);
    long bn = t / 8;
    int n = (int)(bn % Nn), b = (int)(bn / Nn);
    o[t] = lg[((long)b * 8 + c) * Nn + n];
}

// ---------------------------------------------------------------------------
// Host orchestration
// ---------------------------------------------------------------------------
extern "C" void kernel_launch(void* const* d_in, const int* in_sizes, int n_in,
                              void* d_out, int out_size, void* d_ws, size_t ws_size,
                              hipStream_t stream) {
    (void)in_sizes; (void)n_in; (void)out_size; (void)ws_size;
    const int B = 16, N0 = 2048;
    auto IN = [&](int i) { return (const float*)d_in[i]; };

    char* base = (char*)d_ws;
    size_t off = 0;
    auto allocf = [&](size_t n) -> float* {
        float* p = (float*)(base + off);
        off += ((n * sizeof(float) + 255) & ~(size_t)255);
        return p;
    };
    auto alloci = [&](size_t n) -> int* {
        int* p = (int*)(base + off);
        off += ((n * sizeof(int) + 255) & ~(size_t)255);
        return p;
    };

    float* h0     = allocf((size_t)B * 3 * N0);
    float* tmp    = allocf((size_t)B * 1024 * 256);       // largest pre-BN conv output
    float* feat0  = allocf((size_t)B * 64 * N0);
    float* feat1  = allocf((size_t)B * 128 * 1024);
    float* feat2  = allocf((size_t)B * 256 * 512);
    float* feat3  = allocf((size_t)B * 512 * 256);
    float* xyz1   = allocf((size_t)B * 1024 * 3);
    float* xyz2   = allocf((size_t)B * 512 * 3);
    float* xyz3   = allocf((size_t)B * 256 * 3);
    int*   fps1   = alloci((size_t)B * 1024);
    int*   fps2   = alloci((size_t)B * 512);
    int*   fps3   = alloci((size_t)B * 256);
    int*   knn1   = alloci((size_t)B * 1024 * 32);
    int*   knn2   = alloci((size_t)B * 512 * 32);
    int*   knn3b  = alloci((size_t)B * 256 * 32);
    float* statsA = allocf(2 * 512);
    float* statsB = allocf(2 * 512);
    float* statsG = allocf(2 * 4096);
    float* hA     = allocf((size_t)B * 512 * 256);
    float* hB     = allocf((size_t)B * 512 * 256);
    float* xqk    = allocf((size_t)B * 128 * 256);
    float* xv     = allocf((size_t)B * 512 * 256);
    float* attn   = allocf((size_t)B * 256 * 256);
    float* xr     = allocf((size_t)B * 512 * 256);
    float* dif    = allocf((size_t)B * 512 * 256);
    float* outs   = allocf((size_t)B * 2048 * 256);
    float* hmaxb  = allocf((size_t)B * 2048);
    float* fusein = allocf((size_t)B * 4096 * 256);
    float* fuseo  = allocf((size_t)B * 1024 * 256);
    int*   nn3i   = alloci((size_t)B * 2048 * 3);
    float* nn3w   = allocf((size_t)B * 2048 * 3);
    float* cat    = allocf((size_t)B * 1280 * 512);
    float* fo3    = allocf((size_t)B * 512 * 512);
    float* fo2    = allocf((size_t)B * 256 * 1024);
    float* fo1    = allocf((size_t)B * 128 * 2048);

    auto gemmW = [&](const float* W, const float* X, const float* bias, float* Y, int M, int Nn, int K) {
        dim3 g((Nn + 63) / 64, (M + 63) / 64, B);
        gemm_wmma<false, false><<<g, 256, 0, stream>>>(W, X, bias, Y, M, Nn, K, K, 0);
    };
    auto bnseq = [&](const float* xin, float* yout, const float* g_, const float* bt, const float* res,
                     int C, int Nn, int act) {
        bn_stats<<<C, 256, 0, stream>>>(xin, statsG, C, B, Nn);
        long total = (long)B * C * Nn;
        bn_apply<<<(int)((total + 255) / 256), 256, 0, stream>>>(xin, yout, statsG, g_, bt, res, C, Nn,
                                                                 total, act, 1.0f / (float)(B * Nn));
    };

    // ---------------- stem: conv1 (3->64) + conv2 (64->64) ------------------
    {
        long tot = (long)B * 3 * N0;
        transpose_in<<<(int)((tot + 255) / 256), 256, 0, stream>>>(IN(0), h0, N0, tot);
    }
    gemmW(IN(1), h0, nullptr, tmp, 64, N0, 3);
    bnseq(tmp, feat0, IN(2), IN(3), nullptr, 64, N0, 1);
    gemmW(IN(4), feat0, nullptr, tmp, 64, N0, 64);
    bnseq(tmp, feat0, IN(5), IN(6), nullptr, 64, N0, 1);

    // ------------- sample_and_group + local_op levels -----------------------
    auto level = [&](const float* xyzsrc, int nsrc, int np, const float* featsrc, int d,
                     int pbase, int* fpsb, int* knnb, float* xyznew, float* featout) {
        int c = 2 * d;
        fps_kernel<<<B, 256, 0, stream>>>(xyzsrc, fpsb, nsrc, np);
        int tot = B * np;
        gather_xyz<<<(tot + 255) / 256, 256, 0, stream>>>(xyzsrc, fpsb, xyznew, nsrc, np, tot);
        knn32_kernel<<<dim3(np, B), 256, 0, stream>>>(xyznew, xyzsrc, knnb, nsrc);
        hipMemsetAsync(statsA, 0, 2 * c * sizeof(float), stream);
        hipMemsetAsync(statsB, 0, 2 * c * sizeof(float), stream);
        float invc = 1.0f / (float)((long)B * np * 32);
        dim3 lg(np, B);
        local_pass<0><<<lg, 256, 0, stream>>>(featsrc, fpsb, knnb, IN(pbase), IN(pbase + 3), statsA, statsB,
                                              IN(pbase + 1), IN(pbase + 2), IN(pbase + 4), IN(pbase + 5),
                                              featout, d, c, np, nsrc, invc);
        local_pass<1><<<lg, 256, 0, stream>>>(featsrc, fpsb, knnb, IN(pbase), IN(pbase + 3), statsA, statsB,
                                              IN(pbase + 1), IN(pbase + 2), IN(pbase + 4), IN(pbase + 5),
                                              featout, d, c, np, nsrc, invc);
        local_pass<2><<<lg, 256, 0, stream>>>(featsrc, fpsb, knnb, IN(pbase), IN(pbase + 3), statsA, statsB,
                                              IN(pbase + 1), IN(pbase + 2), IN(pbase + 4), IN(pbase + 5),
                                              featout, d, c, np, nsrc, invc);
    };
    level(IN(0), 2048, 1024, feat0, 64,  7,  fps1, knn1,  xyz1, feat1);
    level(xyz1,  1024, 512,  feat1, 128, 13, fps2, knn2,  xyz2, feat2);
    level(xyz2,  512,  256,  feat2, 256, 19, fps3, knn3b, xyz3, feat3);

    // ---------------- point-transformer trunk ------------------------------
    gemmW(IN(25), feat3, nullptr, tmp, 512, 256, 512);
    bnseq(tmp, hA, IN(26), IN(27), nullptr, 512, 256, 1);
    gemmW(IN(28), hA, nullptr, tmp, 512, 256, 512);
    bnseq(tmp, hB, IN(29), IN(30), nullptr, 512, 256, 1);

    float* cur = hB;
    float* oth = hA;
    long hTotal = (long)B * 512 * 256;
    for (int i = 0; i < 4; ++i) {
        int pb = 31 + i * 7;  // qk_w, v_w, v_b, t_w, t_b, bn_g, bn_b
        gemmW(IN(pb + 0), cur, nullptr, xqk, 128, 256, 512);
        gemmW(IN(pb + 1), cur, IN(pb + 2), xv, 512, 256, 512);
        // energy[b,n,m] = sum_c xqk[b,c,n] * xqk[b,c,m]  (A^T, per-batch)
        {
            dim3 g((256 + 63) / 64, (256 + 63) / 64, B);
            gemm_wmma<true, true><<<g, 256, 0, stream>>>(xqk, xqk, nullptr, attn, 256, 256, 128, 256,
                                                         (long)128 * 256);
        }
        softmax_rows<<<B * 256, 256, 0, stream>>>(attn, 256);
        colnorm<<<B * 256, 256, 0, stream>>>(attn, 256);
        // xr[b,c,m] = sum_n xv[b,c,n] * attn[b,n,m]  (per-batch NN)
        {
            dim3 g((256 + 63) / 64, (512 + 63) / 64, B);
            gemm_wmma<true, false><<<g, 256, 0, stream>>>(xv, attn, nullptr, xr, 512, 256, 256, 256,
                                                          (long)512 * 256);
        }
        vsub<<<(int)((hTotal + 255) / 256), 256, 0, stream>>>(cur, xr, dif, hTotal);
        gemmW(IN(pb + 3), dif, IN(pb + 4), tmp, 512, 256, 512);
        bnseq(tmp, oth, IN(pb + 5), IN(pb + 6), cur, 512, 256, 1);
        copy_rows<<<(int)((hTotal + 255) / 256), 256, 0, stream>>>(oth, outs, i * 512, 512, 2048, 256, hTotal);
        float* t = cur; cur = oth; oth = t;
    }

    // ---------------- fuse conv over [outs ; broadcast max] ----------------
    maxn_kernel<<<(B * 2048 + 255) / 256, 256, 0, stream>>>(outs, hmaxb, 256, B * 2048);
    {
        long tot = (long)B * 4096 * 256;
        buildfuse<<<(int)((tot + 255) / 256), 256, 0, stream>>>(outs, hmaxb, fusein, 256, tot);
    }
    gemmW(IN(59), fusein, nullptr, tmp, 1024, 256, 4096);
    bnseq(tmp, fuseo, IN(60), IN(61), nullptr, 1024, 256, 2);  // leaky 0.2

    // ---------------- feature propagation chain ----------------------------
    auto fprop = [&](const float* xyzd, const float* xyzs, int nd, int ns,
                     const float* p1, int C1, const float* p2, int C2,
                     int widx, float* outbuf, int Cout) {
        int tot = B * nd;
        knn3w<<<(tot + 255) / 256, 256, 0, stream>>>(xyzd, xyzs, nn3i, nn3w, tot, nd, ns);
        long ctot = (long)B * (C1 + C2) * nd;
        fp_concat<<<(int)((ctot + 255) / 256), 256, 0, stream>>>(p1, p2, nn3i, nn3w, cat, C1, C2, nd, ns, ctot);
        gemmW(IN(widx), cat, nullptr, tmp, Cout, nd, C1 + C2);
        bnseq(tmp, outbuf, IN(widx + 1), IN(widx + 2), nullptr, Cout, nd, 1);
    };
    fprop(xyz2,  xyz3, 512,  256,  feat2, 256, fuseo, 1024, 62, fo3, 512);
    fprop(xyz1,  xyz2, 1024, 512,  feat1, 128, fo3,   512,  65, fo2, 256);
    fprop(IN(0), xyz1, 2048, 1024, feat0, 64,  fo2,   256,  68, fo1, 128);

    // ---------------- logits + output transpose ----------------------------
    gemmW(IN(71), fo1, IN(72), tmp, 8, N0, 128);
    {
        long tot = (long)B * N0 * 8;
        transpose_out<<<(int)((tot + 255) / 256), 256, 0, stream>>>(tmp, (float*)d_out, N0, tot);
    }
}
